// SSMLayerV2_29703993819796
// MI455X (gfx1250) — matrix operand order repaired and checked by reading, not compiled, not run on hardware
//
#include <hip/hip_runtime.h>
#include <hip/hip_bf16.h>
#include <math.h>

// ---------------------------------------------------------------------------
// Types for CDNA5 WMMA (wave32)
// ---------------------------------------------------------------------------
typedef __bf16 bf16_t;
typedef __attribute__((ext_vector_type(16))) __bf16 v16bf;
typedef __attribute__((ext_vector_type(8)))  __bf16 v8bf;
typedef __attribute__((ext_vector_type(8)))  float  v8f;

#define DEV __device__ __forceinline__

constexpr int Bb = 4, T = 2048, Dm = 1024, Nst = 16;
constexpr int M = Bb * T;            // 8192 token rows
constexpr int TWO_D = 2 * Dm;

DEV bf16_t to_bf16(float f) { return (bf16_t)f; }

DEV float siluf(float x)    { return x / (1.0f + __expf(-x)); }
DEV float sigmoidf_(float x){ return 1.0f / (1.0f + __expf(-x)); }
DEV float softplusf(float x){ return (x > 20.0f) ? x : log1pf(__expf(x)); }

// ---------------------------------------------------------------------------
// WMMA fragment helpers (row/col/C layout per CDNA5 ISA 7.12.2; k order is this code's contiguous one, same in A and B)
// A 16x32 bf16: lane l -> row = l&15; K = (l>>4)*16 + i,
//   loaded as two contiguous 16B halves per lane.
// B 32x16 bf16 (B[k,n]=W[n,k], W row-major over k): lane l -> col=l&15,
//   K = (l>>4)*16 + i -> one contiguous 32B run per lane.
// C/D 16x16 f32: lane l col=l&15, vgpr v -> row = v + 8*(l>>4).
// ---------------------------------------------------------------------------
DEV v16bf lda16(const bf16_t* __restrict__ p) {
  v8bf lo = *(const v8bf*)p;
  v8bf hi = *(const v8bf*)(p + 8);
  return __builtin_shufflevector(lo, hi, 0,1,2,3,4,5,6,7,8,9,10,11,12,13,14,15);
}
DEV v16bf ldb16(const bf16_t* __restrict__ p) { return *(const v16bf*)p; }

DEV v8f wmma_bf16(v16bf a, v16bf b, v8f c) {
  return __builtin_amdgcn_wmma_f32_16x16x32_bf16(
      /*neg_a=*/false, a, /*neg_b=*/false, b,
      /*c_mod=*/(short)0, c, /*reuse_a=*/false, /*reuse_b=*/false);
}

// butterfly sum over the 16 lanes of a wave32 half (masks stay inside half)
DEV float half16_allreduce(float v) {
  v += __shfl_xor(v, 1, 32);
  v += __shfl_xor(v, 2, 32);
  v += __shfl_xor(v, 4, 32);
  v += __shfl_xor(v, 8, 32);
  return v;
}

// ---------------------------------------------------------------------------
// Fully-unrolled ping-pong 32x64 wave-tile GEMM core (K = Dm = 1024).
// K-loop fully unrolled: all load addresses fold into 24-bit immediate
// offsets off 6 fixed base registers -> no per-iteration address VALU,
// no WMMA->VALU hazard NOPs, fine-grained s_wait_loadcnt scheduling.
// Two named fragment sets alternate as load targets (no rotation movs).
// ---------------------------------------------------------------------------
DEV void gemm_core_32x64(const bf16_t* __restrict__ A, const bf16_t* __restrict__ W,
                         int row0, int col0, int lane, v8f acc[2][4]) {
  constexpr int K = Dm;
  const size_t aoff0 = (size_t)(row0 + (lane & 15)) * K + ((lane >> 4) << 4);
  const size_t aoff1 = aoff0 + (size_t)16 * K;
  size_t boff[4];
  #pragma unroll
  for (int j = 0; j < 4; ++j)
    boff[j] = (size_t)(col0 + 16 * j + (lane & 15)) * K + ((lane >> 4) << 4);

  v16bf a0[2], b0[4], a1[2], b1[4];
  a0[0] = lda16(A + aoff0);
  a0[1] = lda16(A + aoff1);
  #pragma unroll
  for (int j = 0; j < 4; ++j) b0[j] = ldb16(W + boff[j]);

  #pragma unroll
  for (int kb = 0; kb < K; kb += 64) {
    int k1 = kb + 32;
    if (kb + 128 < K) {
      __builtin_prefetch(A + aoff0 + kb + 128, 0, 1);
      __builtin_prefetch(A + aoff1 + kb + 128, 0, 1);
    }
    a1[0] = lda16(A + aoff0 + k1);
    a1[1] = lda16(A + aoff1 + k1);
    #pragma unroll
    for (int j = 0; j < 4; ++j) b1[j] = ldb16(W + boff[j] + k1);

    #pragma unroll
    for (int j = 0; j < 4; ++j) {
      acc[0][j] = wmma_bf16(a0[0], b0[j], acc[0][j]);
      acc[1][j] = wmma_bf16(a0[1], b0[j], acc[1][j]);
    }

    int k2 = (kb + 64 < K) ? (kb + 64) : 0;   // tail reloads k=0 into dead buffer
    a0[0] = lda16(A + aoff0 + k2);
    a0[1] = lda16(A + aoff1 + k2);
    #pragma unroll
    for (int j = 0; j < 4; ++j) b0[j] = ldb16(W + boff[j] + k2);

    #pragma unroll
    for (int j = 0; j < 4; ++j) {
      acc[0][j] = wmma_bf16(a1[0], b1[j], acc[0][j]);
      acc[1][j] = wmma_bf16(a1[1], b1[j], acc[1][j]);
    }
  }
}

// ---------------------------------------------------------------------------
// fp32 -> bf16 weight conversion (strided source rows)
// ---------------------------------------------------------------------------
__global__ void convert_bf16_kernel(const float* __restrict__ src, bf16_t* __restrict__ dst,
                                    int rows, int src_stride, int ncols) {
  size_t total = (size_t)rows * ncols;
  for (size_t i = (size_t)blockIdx.x * blockDim.x + threadIdx.x; i < total;
       i += (size_t)gridDim.x * blockDim.x) {
    size_t r = i / ncols, c = i % ncols;
    dst[r * ncols + c] = to_bf16(src[r * src_stride + c]);
  }
}

// ---------------------------------------------------------------------------
// LayerNorm over D=1024, output bf16. One block per token row.
// ---------------------------------------------------------------------------
__global__ __launch_bounds__(256) void layernorm_bf16_kernel(
    const float* __restrict__ x, const float* __restrict__ g, const float* __restrict__ b,
    bf16_t* __restrict__ out) {
  int row = blockIdx.x;
  const float* xr = x + (size_t)row * Dm;
  int lane = threadIdx.x & 31, wave = threadIdx.x >> 5;

  float lsum = 0.0f, lsq = 0.0f;
  #pragma unroll
  for (int i = threadIdx.x; i < Dm; i += 256) {
    float v = xr[i];
    lsum += v; lsq += v * v;
  }
  #pragma unroll
  for (int m = 1; m < 32; m <<= 1) {
    lsum += __shfl_xor(lsum, m, 32);
    lsq  += __shfl_xor(lsq,  m, 32);
  }
  __shared__ float s1[8], s2[8];
  if (lane == 0) { s1[wave] = lsum; s2[wave] = lsq; }
  __syncthreads();
  float tsum = 0.0f, tsq = 0.0f;
  #pragma unroll
  for (int w = 0; w < 8; ++w) { tsum += s1[w]; tsq += s2[w]; }
  float mean = tsum / Dm;
  float var  = tsq / Dm - mean * mean;
  float inv  = rsqrtf(var + 1e-5f);
  #pragma unroll
  for (int i = threadIdx.x; i < Dm; i += 256)
    out[(size_t)row * Dm + i] = to_bf16((xr[i] - mean) * inv * g[i] + b[i]);
}

// ---------------------------------------------------------------------------
// GEMM 1: xg = xn @ W_in^T  (M x 2048, K=1024)
// Block: 8 waves, 2(M)x4(N); wave tile 32x64; block tile 64x256.
// Epilogue: cols < D -> silu -> x_ssm (bf16) + row-sum atomics (x_sc)
//           cols >= D -> sigmoid -> z (f32)
// ---------------------------------------------------------------------------
__global__ __launch_bounds__(256) void gemm_win_kernel(
    const bf16_t* __restrict__ xn, const bf16_t* __restrict__ Wbf,
    bf16_t* __restrict__ xs, float* __restrict__ z, float* __restrict__ xsc_sum) {
  int wave = threadIdx.x >> 5, lane = threadIdx.x & 31;
  int row0 = blockIdx.x * 64 + (wave & 1) * 32;
  int col0 = blockIdx.y * 256 + (wave >> 1) * 64;

  v8f acc[2][4] = {{v8f{}, v8f{}, v8f{}, v8f{}}, {v8f{}, v8f{}, v8f{}, v8f{}}};
  gemm_core_32x64(xn, Wbf, row0, col0, lane, acc);

  int half = lane >> 4, cn = lane & 15;
  bool is_x = (col0 < Dm);
  float rowsum[2][8] = {};
  #pragma unroll
  for (int i = 0; i < 2; ++i) {
    #pragma unroll
    for (int j = 0; j < 4; ++j) {
      int col = col0 + 16 * j + cn;
      #pragma unroll
      for (int v = 0; v < 8; ++v) {
        int row = row0 + 16 * i + v + 8 * half;
        float val = acc[i][j][v];
        if (is_x) {
          float s = siluf(val);
          xs[(size_t)row * Dm + col] = to_bf16(s);
          rowsum[i][v] += s;
        } else {
          z[(size_t)row * Dm + (col - Dm)] = sigmoidf_(val);
        }
      }
    }
  }
  if (is_x) {
    #pragma unroll
    for (int i = 0; i < 2; ++i)
      #pragma unroll
      for (int v = 0; v < 8; ++v) {
        float s = half16_allreduce(rowsum[i][v]);
        if (cn == 0) atomicAdd(&xsc_sum[row0 + 16 * i + v + 8 * half], s);
      }
  }
}

// ---------------------------------------------------------------------------
// Dual projection: B_t / C_t = l2norm(x_ssm @ W^T), N=16, K=1024.
// One wave per 32-row stripe; 2 A frags shared by both accumulators.
// ---------------------------------------------------------------------------
__global__ __launch_bounds__(256) void proj_bc_kernel(
    const bf16_t* __restrict__ xs, const bf16_t* __restrict__ WBbf,
    const bf16_t* __restrict__ WCbf, float* __restrict__ Bt, float* __restrict__ Ct) {
  constexpr int K = Dm;
  int wave = threadIdx.x >> 5, lane = threadIdx.x & 31;
  int row0 = (blockIdx.x * 8 + wave) * 32;

  const size_t aoff0 = (size_t)(row0 + (lane & 15)) * K + ((lane >> 4) << 4);
  const size_t aoff1 = aoff0 + (size_t)16 * K;
  const size_t boff  = (size_t)(lane & 15) * K + ((lane >> 4) << 4);

  v8f accB[2] = {v8f{}, v8f{}}, accC[2] = {v8f{}, v8f{}};
  #pragma unroll
  for (int kb = 0; kb < K; kb += 32) {
    v16bf a0 = lda16(xs + aoff0 + kb), a1 = lda16(xs + aoff1 + kb);
    v16bf vb = ldb16(WBbf + boff + kb), vc = ldb16(WCbf + boff + kb);
    accB[0] = wmma_bf16(a0, vb, accB[0]);
    accB[1] = wmma_bf16(a1, vb, accB[1]);
    accC[0] = wmma_bf16(a0, vc, accC[0]);
    accC[1] = wmma_bf16(a1, vc, accC[1]);
  }
  int half = lane >> 4, cn = lane & 15;
  #pragma unroll
  for (int i = 0; i < 2; ++i)
    #pragma unroll
    for (int v = 0; v < 8; ++v) {
      int row = row0 + 16 * i + v + 8 * half;
      float bv = accB[i][v], cv = accC[i][v];
      float nb = half16_allreduce(bv * bv);
      float nc = half16_allreduce(cv * cv);
      Bt[(size_t)row * 16 + cn] = bv / fmaxf(sqrtf(nb), 1e-12f);
      Ct[(size_t)row * 16 + cn] = cv / fmaxf(sqrtf(nc), 1e-12f);
    }
}

// ---------------------------------------------------------------------------
// dt GEMM: only dt_sum (row sums of softplus) is needed.
// val = softplus(x_ssm@W_dt[:, :D]^T + tc * W_dt[:, D] + b_dt)
// ---------------------------------------------------------------------------
__global__ __launch_bounds__(256) void gemm_dt_kernel(
    const bf16_t* __restrict__ xs, const bf16_t* __restrict__ Wdt_bf,
    const float* __restrict__ Wdt_f32, const float* __restrict__ b_dt,
    const float* __restrict__ tc, float* __restrict__ dt_sum) {
  constexpr int Dp1 = Dm + 1;
  int wave = threadIdx.x >> 5, lane = threadIdx.x & 31;
  int row0 = blockIdx.x * 64 + (wave & 1) * 32;
  int col0 = blockIdx.y * 256 + (wave >> 1) * 64;

  v8f acc[2][4] = {{v8f{}, v8f{}, v8f{}, v8f{}}, {v8f{}, v8f{}, v8f{}, v8f{}}};
  gemm_core_32x64(xs, Wdt_bf, row0, col0, lane, acc);

  int half = lane >> 4, cn = lane & 15;
  float rowsum[2][8] = {};
  #pragma unroll
  for (int j = 0; j < 4; ++j) {
    int col = col0 + 16 * j + cn;
    float wl = Wdt_f32[(size_t)col * Dp1 + (Dp1 - 1)];
    float bb = b_dt[col];
    #pragma unroll
    for (int i = 0; i < 2; ++i)
      #pragma unroll
      for (int v = 0; v < 8; ++v) {
        float tcv = tc[row0 + 16 * i + v + 8 * half];
        rowsum[i][v] += softplusf(acc[i][j][v] + tcv * wl + bb);
      }
  }
  #pragma unroll
  for (int i = 0; i < 2; ++i)
    #pragma unroll
    for (int v = 0; v < 8; ++v) {
      float s = half16_allreduce(rowsum[i][v]);
      if (cn == 0) atomicAdd(&dt_sum[row0 + 16 * i + v + 8 * half], s);
    }
}

// ---------------------------------------------------------------------------
// e GEMM: e = ln2(engram_v) @ W_eng^T ; store e (bf16) + row sums of e*e.
// ---------------------------------------------------------------------------
__global__ __launch_bounds__(256) void gemm_eng_kernel(
    const bf16_t* __restrict__ en, const bf16_t* __restrict__ Wbf,
    bf16_t* __restrict__ e_out, float* __restrict__ e2sum) {
  int wave = threadIdx.x >> 5, lane = threadIdx.x & 31;
  int row0 = blockIdx.x * 64 + (wave & 1) * 32;
  int col0 = blockIdx.y * 256 + (wave >> 1) * 64;

  v8f acc[2][4] = {{v8f{}, v8f{}, v8f{}, v8f{}}, {v8f{}, v8f{}, v8f{}, v8f{}}};
  gemm_core_32x64(en, Wbf, row0, col0, lane, acc);

  int half = lane >> 4, cn = lane & 15;
  float rowsum[2][8] = {};
  #pragma unroll
  for (int i = 0; i < 2; ++i)
    #pragma unroll
    for (int j = 0; j < 4; ++j) {
      int col = col0 + 16 * j + cn;
      #pragma unroll
      for (int v = 0; v < 8; ++v) {
        int row = row0 + 16 * i + v + 8 * half;
        float val = acc[i][j][v];
        e_out[(size_t)row * Dm + col] = to_bf16(val);
        rowsum[i][v] += val * val;
      }
    }
  #pragma unroll
  for (int i = 0; i < 2; ++i)
    #pragma unroll
    for (int v = 0; v < 8; ++v) {
      float s = half16_allreduce(rowsum[i][v]);
      if (cn == 0) atomicAdd(&e2sum[row0 + 16 * i + v + 8 * half], s);
    }
}

// ---------------------------------------------------------------------------
// Sequential SSM scan. 1 block, 64 threads: thread -> (b = t>>4, n = t&15).
// ---------------------------------------------------------------------------
__global__ void scan_kernel(const float* __restrict__ Bt, const float* __restrict__ Ct,
                            const float* __restrict__ dt_sum, const float* __restrict__ xsc_sum,
                            const float* __restrict__ log_A, float* __restrict__ y_ssm) {
  int t = threadIdx.x;
  int b = t >> 4, n = t & 15;
  float la = fminf(fmaxf(log_A[n], -10.0f), -0.5f);
  float A = -__expf(la);
  float invD = 1.0f / (float)Dm;
  float h = 0.0f;
  for (int tt = 0; tt < T; ++tt) {
    int row = b * T + tt;
    float dts = fminf(fmaxf(dt_sum[row] * invD, 1e-3f), 5.0f);
    float ab = __expf(dts * A);
    if (!(ab == ab)) ab = 0.1f;                    // NaN guard per reference
    float bbar = (1.0f - ab) / (A - 1e-8f) * Bt[(size_t)row * 16 + n];
    float bx = bbar * (xsc_sum[row] * invD);
    h = fminf(fmaxf(ab * h + bx, -10.0f), 10.0f);
    float y = Ct[(size_t)row * 16 + n] * h;
    y = half16_allreduce(y);
    if (n == 0) y_ssm[row] = y;
  }
}

// ---------------------------------------------------------------------------
// Combine: y = y_ssm * z + sigmoid(e2sum/D) * e   -> bf16 for final GEMM.
// ---------------------------------------------------------------------------
__global__ void combine_kernel(const float* __restrict__ z, const bf16_t* __restrict__ e,
                               const float* __restrict__ e2sum, const float* __restrict__ y_ssm,
                               bf16_t* __restrict__ yb) {
  size_t total = (size_t)M * Dm;
  for (size_t i = (size_t)blockIdx.x * blockDim.x + threadIdx.x; i < total;
       i += (size_t)gridDim.x * blockDim.x) {
    size_t row = i / Dm;
    float g = sigmoidf_(e2sum[row] / (float)Dm);
    float val = y_ssm[row] * z[i] + g * (float)e[i];
    yb[i] = to_bf16(val);
  }
}

// ---------------------------------------------------------------------------
// Final GEMM: out = y @ W_out^T + residual(x), fp32 output.
// ---------------------------------------------------------------------------
__global__ __launch_bounds__(256) void gemm_out_kernel(
    const bf16_t* __restrict__ yb, const bf16_t* __restrict__ Wbf,
    const float* __restrict__ res, float* __restrict__ out) {
  int wave = threadIdx.x >> 5, lane = threadIdx.x & 31;
  int row0 = blockIdx.x * 64 + (wave & 1) * 32;
  int col0 = blockIdx.y * 256 + (wave >> 1) * 64;

  v8f acc[2][4] = {{v8f{}, v8f{}, v8f{}, v8f{}}, {v8f{}, v8f{}, v8f{}, v8f{}}};
  gemm_core_32x64(yb, Wbf, row0, col0, lane, acc);

  int half = lane >> 4, cn = lane & 15;
  #pragma unroll
  for (int i = 0; i < 2; ++i)
    #pragma unroll
    for (int j = 0; j < 4; ++j) {
      int col = col0 + 16 * j + cn;
      #pragma unroll
      for (int v = 0; v < 8; ++v) {
        size_t row = (size_t)(row0 + 16 * i + v + 8 * half);
        out[row * Dm + col] = acc[i][j][v] + res[row * Dm + col];
      }
    }
}

// ---------------------------------------------------------------------------
// Host side
// ---------------------------------------------------------------------------
extern "C" void kernel_launch(void* const* d_in, const int* in_sizes, int n_in,
                              void* d_out, int out_size, void* d_ws, size_t ws_size,
                              hipStream_t stream) {
  (void)in_sizes; (void)n_in; (void)out_size; (void)ws_size;
  const float* x      = (const float*)d_in[0];
  const float* engram = (const float*)d_in[1];
  const float* tc     = (const float*)d_in[2];
  const float* W_in   = (const float*)d_in[3];
  const float* W_out  = (const float*)d_in[4];
  const float* W_B    = (const float*)d_in[5];
  const float* W_C    = (const float*)d_in[6];
  const float* W_dt   = (const float*)d_in[7];
  const float* b_dt   = (const float*)d_in[8];
  const float* log_A  = (const float*)d_in[9];
  const float* W_eng  = (const float*)d_in[10];
  const float* ln1_g  = (const float*)d_in[11];
  const float* ln1_b  = (const float*)d_in[12];
  const float* ln2_g  = (const float*)d_in[13];
  const float* ln2_b  = (const float*)d_in[14];
  float* out = (float*)d_out;

  // ---- carve workspace ----
  char* p = (char*)d_ws;
  auto carve = [&](size_t bytes) -> void* {
    void* r = (void*)p;
    p += (bytes + 255) & ~(size_t)255;
    return r;
  };
  bf16_t* xn_bf   = (bf16_t*)carve((size_t)M * Dm * 2);
  bf16_t* en_bf   = (bf16_t*)carve((size_t)M * Dm * 2);
  bf16_t* xs_bf   = (bf16_t*)carve((size_t)M * Dm * 2);
  bf16_t* e_bf    = (bf16_t*)carve((size_t)M * Dm * 2);
  bf16_t* yb_bf   = (bf16_t*)carve((size_t)M * Dm * 2);
  float*  z_f     = (float*) carve((size_t)M * Dm * 4);
  bf16_t* Win_bf  = (bf16_t*)carve((size_t)TWO_D * Dm * 2);
  bf16_t* Wdt_bf  = (bf16_t*)carve((size_t)Dm * Dm * 2);
  bf16_t* Weng_bf = (bf16_t*)carve((size_t)Dm * Dm * 2);
  bf16_t* Wout_bf = (bf16_t*)carve((size_t)Dm * Dm * 2);
  bf16_t* WB_bf   = (bf16_t*)carve((size_t)Nst * Dm * 2);
  bf16_t* WC_bf   = (bf16_t*)carve((size_t)Nst * Dm * 2);
  float*  Bt      = (float*) carve((size_t)M * Nst * 4);
  float*  Ct      = (float*) carve((size_t)M * Nst * 4);
  float*  xsc_sum = (float*) carve((size_t)M * 4);
  float*  dt_sum  = (float*) carve((size_t)M * 4);
  float*  e2sum   = (float*) carve((size_t)M * 4);
  float*  y_ssm   = (float*) carve((size_t)M * 4);

  // ---- zero reduction accumulators (capture-safe) ----
  hipMemsetAsync(xsc_sum, 0, (size_t)M * 4, stream);
  hipMemsetAsync(dt_sum,  0, (size_t)M * 4, stream);
  hipMemsetAsync(e2sum,   0, (size_t)M * 4, stream);

  // ---- weight conversion fp32 -> bf16 ----
  auto cvt = [&](const float* src, bf16_t* dst, int rows, int stride, int cols) {
    size_t total = (size_t)rows * cols;
    int blocks = (int)((total + 255) / 256);
    if (blocks > 4096) blocks = 4096;
    convert_bf16_kernel<<<blocks, 256, 0, stream>>>(src, dst, rows, stride, cols);
  };
  cvt(W_in,  Win_bf,  TWO_D, Dm,     Dm);
  cvt(W_dt,  Wdt_bf,  Dm,    Dm + 1, Dm);   // drop the last (tc) column
  cvt(W_eng, Weng_bf, Dm,    Dm,     Dm);
  cvt(W_out, Wout_bf, Dm,    Dm,     Dm);
  cvt(W_B,   WB_bf,   Nst,   Dm,     Dm);
  cvt(W_C,   WC_bf,   Nst,   Dm,     Dm);

  // ---- layernorms -> bf16 activations ----
  layernorm_bf16_kernel<<<M, 256, 0, stream>>>(x,      ln1_g, ln1_b, xn_bf);
  layernorm_bf16_kernel<<<M, 256, 0, stream>>>(engram, ln2_g, ln2_b, en_bf);

  // ---- GEMM 1: x_ssm / z (+ x_sc row sums) ----
  {
    dim3 grid(M / 64, TWO_D / 256);
    gemm_win_kernel<<<grid, 256, 0, stream>>>(xn_bf, Win_bf, xs_bf, z_f, xsc_sum);
  }

  // ---- B_t / C_t projections with fused l2norm ----
  proj_bc_kernel<<<M / 256, 256, 0, stream>>>(xs_bf, WB_bf, WC_bf, Bt, Ct);

  // ---- dt row sums (softplus fused) ----
  {
    dim3 grid(M / 64, Dm / 256);
    gemm_dt_kernel<<<grid, 256, 0, stream>>>(xs_bf, Wdt_bf, W_dt, b_dt, tc, dt_sum);
  }

  // ---- engram projection e (+ e*e row sums) ----
  {
    dim3 grid(M / 64, Dm / 256);
    gemm_eng_kernel<<<grid, 256, 0, stream>>>(en_bf, Weng_bf, e_bf, e2sum);
  }

  // ---- sequential scan over T ----
  scan_kernel<<<1, 64, 0, stream>>>(Bt, Ct, dt_sum, xsc_sum, log_A, y_ssm);

  // ---- combine y = y_ssm*z + g*e -> bf16 ----
  combine_kernel<<<4096, 256, 0, stream>>>(z_f, e_bf, e2sum, y_ssm, yb_bf);

  // ---- final GEMM + residual ----
  {
    dim3 grid(M / 64, Dm / 256);
    gemm_out_kernel<<<grid, 256, 0, stream>>>(yb_bf, Wout_bf, x, out);
  }
}
